// QuadDualInputMambaLayer_48936857371342
// MI455X (gfx1250) — matrix operand order repaired and checked by reading, not compiled, not run on hardware
//
#include <hip/hip_runtime.h>

// ---------------- problem constants ----------------
#define BB   2
#define LL   2048
#define DD   512
#define DI   1024
#define DS   16
#define DC   4
#define DTR  32
#define NL   4
#define TT   (2*LL)          // 4096 (fwd + reversed halves scanned as ONE sequence)
#define MM   (BB*TT)         // 8192 rows per layer in every GEMM

typedef __bf16 bf16;
typedef __bf16 v16bf  __attribute__((ext_vector_type(16)));
typedef __bf16 bf16x2 __attribute__((ext_vector_type(2)));
typedef float  v8f    __attribute__((ext_vector_type(8)));

__device__ __forceinline__ bf16 f2bf(float f) {
  unsigned u = __builtin_bit_cast(unsigned, f);
  unsigned r = u + 0x7FFFu + ((u >> 16) & 1u);   // round-to-nearest-even
  unsigned short h = (unsigned short)(r >> 16);
  return __builtin_bit_cast(bf16, h);
}
__device__ __forceinline__ float bf2f(bf16 b) {
  unsigned short h = __builtin_bit_cast(unsigned short, b);
  unsigned u = ((unsigned)h) << 16;
  return __builtin_bit_cast(float, u);
}
__device__ __forceinline__ float silu_f(float x) { return x / (1.f + __expf(-x)); }

// ---- CDNA5 async memory->LDS copy (GLOBAL_LOAD_ASYNC_TO_LDS_B128, ASYNCcnt) ----
// LDS operand: flat shared pointer truncated to 32 bits = wave-relative LDS
// address (ISA 10.2: LDS aperture uses addr[31:0] directly).
__device__ __forceinline__ void async_copy_b128(void* lds, const void* gaddr) {
  unsigned loff = (unsigned)(unsigned long long)lds;
  asm volatile("global_load_async_to_lds_b128 %0, %1, off"
               :: "v"(loff), "v"(gaddr) : "memory");
}
template<int N>
__device__ __forceinline__ void wait_asynccnt() {
  asm volatile("s_wait_asynccnt %0" :: "n"(N) : "memory");
}

// ------------------------------------------------------------------
// Generic tiled WMMA GEMM:  C[l] = A[l] (MxK, row stride lda) * W[l]^T
// W stored (N,K) row-major per layer.  256 threads = 8 waves (4x2 grid),
// each wave computes (BM/4)x(BN/2) via 16x16x32 bf16 WMMA fragments.
// Global->LDS staging via async-to-LDS + LDS double buffering; last K
// iteration peeled so the steady-state loop body is branch-free.
// ------------------------------------------------------------------
template<int BM, int BN, bool C_BF16>
__global__ __launch_bounds__(256) void wmma_gemm(
    const bf16* __restrict__ A, const bf16* __restrict__ W,
    void* __restrict__ Cout, int M, int N, int K, int lda)
{
  constexpr int LDA_S = 40;            // 32 halfs + pad (keeps 16B vector align)
  constexpr int LDB_S = 40;
  constexpr int FM = BM / 64;          // M fragments per wave (waveM grid = 4)
  constexpr int FN = BN / 32;          // N fragments per wave (waveN grid = 2)
  constexpr int AV = (BM * 4) / 256;   // async b128 per thread for A tile
  constexpr int BV = (BN * 4) / 256;   // async b128 per thread for B tile
  constexpr int LPS = AV + BV;         // async instructions per wave per stage
  __shared__ __align__(16) bf16 As[2][BM * LDA_S];
  __shared__ __align__(16) bf16 Bs[2][BN * LDB_S];

  const int tid   = threadIdx.x;
  const int lane  = tid & 31;
  const int wid   = tid >> 5;
  const int waveM = wid & 3;
  const int waveN = wid >> 2;

  const bf16* Ablk = A + (size_t)blockIdx.z * (size_t)M * lda
                       + (size_t)blockIdx.x * BM * lda;
  const bf16* Wblk = W + (size_t)blockIdx.z * (size_t)N * K
                       + (size_t)blockIdx.y * BN * K;

  v8f acc[FM][FN];
  #pragma unroll
  for (int i = 0; i < FM; ++i)
    #pragma unroll
    for (int j = 0; j < FN; ++j)
      #pragma unroll
      for (int e = 0; e < 8; ++e) acc[i][j][e] = 0.f;

  // issue the async stage of one 32-wide K tile into LDS buffer `buf`
  auto stage = [&](int k0, int buf) {
    #pragma unroll
    for (int it = 0; it < AV; ++it) {
      int v = tid + it * 256, row = v >> 2, seg = v & 3;
      async_copy_b128(&As[buf][row * LDA_S + seg * 8],
                      Ablk + (size_t)row * lda + k0 + seg * 8);
    }
    #pragma unroll
    for (int it = 0; it < BV; ++it) {
      int v = tid + it * 256, row = v >> 2, seg = v & 3;
      async_copy_b128(&Bs[buf][row * LDB_S + seg * 8],
                      Wblk + (size_t)row * K + k0 + seg * 8);
    }
  };

  // gather fragments per CDNA5 VGPR layouts and issue FM*FN WMMAs
  auto compute = [&](int buf) {
    v16bf afr[FM], bfr[FN];
    const int mr = lane & 15, kh = lane >> 4;
    #pragma unroll
    for (int fm = 0; fm < FM; ++fm) {
      const int mb = (waveM * FM + fm) * 16 + mr;
      #pragma unroll
      for (int r = 0; r < 8; ++r) {
        int ks = (r >> 2) * 16 + kh * 8 + (r & 3) * 2;   // A: 16x32 bf16 striping
        bf16x2 p = *(const bf16x2*)&As[buf][mb * LDA_S + ks];
        afr[fm][2 * r] = p[0]; afr[fm][2 * r + 1] = p[1];
      }
    }
    #pragma unroll
    for (int fn = 0; fn < FN; ++fn) {
      const int nb = (waveN * FN + fn) * 16 + mr;
      #pragma unroll
      for (int r = 0; r < 8; ++r) {
        int ks = (r >> 2) * 16 + kh * 8 + (r & 3) * 2;   // B: 32x16 bf16 striping (same k order as A)
        bf16x2 p = *(const bf16x2*)&Bs[buf][nb * LDB_S + ks];
        bfr[fn][2 * r] = p[0]; bfr[fn][2 * r + 1] = p[1];
      }
    }
    #pragma unroll
    for (int fm = 0; fm < FM; ++fm)
      #pragma unroll
      for (int fn = 0; fn < FN; ++fn)
        acc[fm][fn] = __builtin_amdgcn_wmma_f32_16x16x32_bf16(
            false, afr[fm], false, bfr[fn], (short)0, acc[fm][fn], false, false);
  };

  const int nk = K / 32;
  stage(0, 0);
  #pragma unroll 1
  for (int kt = 0; kt < nk - 1; ++kt) {
    const int buf = kt & 1;
    // speculative L2 prefetch two tiles ahead (no completion semantics;
    // an over-run address is harmless -> unconditional, branch-free)
    __builtin_prefetch(Ablk + (size_t)(tid >> 2) * lda + (kt + 2) * 32 + (tid & 3) * 8, 0, 3);
    stage((kt + 1) * 32, buf ^ 1);     // next tile in flight behind WMMAs
    wait_asynccnt<LPS>();              // drain only current tile (in-order)
    __syncthreads();
    compute(buf);
    __syncthreads();
  }
  wait_asynccnt<0>();                  // peeled last tile
  __syncthreads();
  compute((nk - 1) & 1);

  // ---- epilogue: C/D layout M = r + 8*(lane>>4), N = lane&15 ----
  const size_t cbase = (size_t)blockIdx.z * (size_t)M * N;
  #pragma unroll
  for (int fm = 0; fm < FM; ++fm) {
    int growb = blockIdx.x * BM + (waveM * FM + fm) * 16 + ((lane >> 4) << 3);
    #pragma unroll
    for (int fn = 0; fn < FN; ++fn) {
      int gcol = blockIdx.y * BN + (waveN * FN + fn) * 16 + (lane & 15);
      #pragma unroll
      for (int r = 0; r < 8; ++r) {
        float v = acc[fm][fn][r];
        size_t off = cbase + (size_t)(growb + r) * N + gcol;
        if (C_BF16) ((bf16*)Cout)[off] = f2bf(v);
        else        ((float*)Cout)[off] = v;
      }
    }
  }
}

// ------------------------------------------------------------------
__global__ void k_f32_to_bf16(const float* __restrict__ in, bf16* __restrict__ out, int n) {
  int i = blockIdx.x * blockDim.x + threadIdx.x;
  if (i < n) out[i] = f2bf(in[i]);
}

// xd (NL,B,TT,D) bf16 from the four fp32 inputs; second half time-reversed.
__global__ void k_build_xd(const float* __restrict__ x0, const float* __restrict__ x1,
                           const float* __restrict__ x2, const float* __restrict__ x3,
                           bf16* __restrict__ xd) {
  int idx = blockIdx.x * blockDim.x + threadIdx.x;
  if (idx >= NL * BB * TT * DD) return;
  int d = idx % DD;
  int t = (idx / DD) % TT;
  int b = (idx / (DD * TT)) % BB;
  int l = idx / (DD * TT * BB);
  const float* X = (l == 0) ? x0 : (l == 1) ? x1 : (l == 2) ? x2 : x3;
  int ts = (t < LL) ? t : (TT - 1 - t);
  xd[idx] = f2bf(X[((size_t)b * LL + ts) * DD + d]);
}

// causal depthwise conv(4) + bias + SiLU over xs = xz[..., :DI]
__global__ void k_conv_silu(const bf16* __restrict__ xz, const float* __restrict__ cw,
                            const float* __restrict__ cb, bf16* __restrict__ xc) {
  long idx = (long)blockIdx.x * blockDim.x + threadIdx.x;
  if (idx >= (long)NL * BB * TT * DI) return;
  int  i  = (int)(idx % DI);
  int  t  = (int)((idx / DI) % TT);
  long lb = idx / ((long)DI * TT);
  int  l  = (int)(lb / BB);
  float acc = cb[l * DI + i];
  const bf16* base = xz + lb * (long)TT * (2 * DI) + i;
  #pragma unroll
  for (int j = 0; j < DC; ++j) {
    int tt = t + j - (DC - 1);
    if (tt >= 0) acc += bf2f(base[(long)tt * (2 * DI)]) * cw[(l * DI + i) * DC + j];
  }
  xc[idx] = f2bf(silu_f(acc));
}

// first DTR cols of x_dbl -> contiguous bf16 A operand for dt_proj GEMM
__global__ void k_slice_dt(const float* __restrict__ x_dbl, bf16* __restrict__ xa, int n) {
  int idx = blockIdx.x * blockDim.x + threadIdx.x;
  if (idx >= n) return;
  int c = idx % DTR, r = idx / DTR;
  xa[idx] = f2bf(x_dbl[(size_t)r * (DTR + 2 * DS) + c]);
}

// selective scan: thread = one channel; 16-state in registers; B/C in LDS (2-slot)
__global__ __launch_bounds__(64) void k_scan(
    const float* __restrict__ dtin, const float* __restrict__ dtb,
    const float* __restrict__ x_dbl, const bf16* __restrict__ xc,
    const bf16* __restrict__ xz, const float* __restrict__ A_log,
    const float* __restrict__ Dp, bf16* __restrict__ y)
{
  const int lb = blockIdx.x;                 // l*BB + b
  const int l  = lb / BB;
  const int i  = blockIdx.y * 64 + threadIdx.x;
  float A[DS], h[DS];
  #pragma unroll
  for (int s = 0; s < DS; ++s) {
    A[s] = -__expf(A_log[((size_t)l * DI + i) * DS + s]);
    h[s] = 0.f;
  }
  const float db  = dtb[l * DI + i];
  const float dpr = Dp [l * DI + i];
  const size_t bch  = (size_t)lb * TT * DI;
  const size_t bdbl = (size_t)lb * TT * (DTR + 2 * DS);
  const size_t bz   = (size_t)lb * TT * (2 * DI);
  __shared__ float BC[2][2 * DS];
  for (int t = 0; t < TT; ++t) {
    if (threadIdx.x < 2 * DS)
      BC[t & 1][threadIdx.x] = x_dbl[bdbl + (size_t)t * (DTR + 2 * DS) + DTR + threadIdx.x];
    __syncthreads();
    float dv = dtin[bch + (size_t)t * DI + i] + db;
    dv = (dv > 20.f) ? dv : log1pf(__expf(dv));              // softplus
    float xv = bf2f(xc[bch + (size_t)t * DI + i]);
    float dx = dv * xv;
    const float* Bt = BC[t & 1];
    const float* Ct = Bt + DS;
    float yv = 0.f;
    #pragma unroll
    for (int s = 0; s < DS; ++s) {
      h[s] = __expf(dv * A[s]) * h[s] + dx * Bt[s];
      yv  += h[s] * Ct[s];
    }
    yv += xv * dpr;
    float zv = bf2f(xz[bz + (size_t)t * (2 * DI) + DI + i]);
    yv *= silu_f(zv);
    y[bch + (size_t)t * DI + i] = f2bf(yv);
  }
}

// 0.5*(fwd + reversed bwd) -> LayerNorm(D) -> + residual input -> d_out
__global__ __launch_bounds__(256) void k_combine_ln(
    const float* __restrict__ ob,
    const float* __restrict__ x0, const float* __restrict__ x1,
    const float* __restrict__ x2, const float* __restrict__ x3,
    const float* __restrict__ g, const float* __restrict__ be,
    float* __restrict__ out)
{
  const int blk = blockIdx.x;                // l*BB*LL + b*LL + t
  const int t = blk % LL;
  const int b = (blk / LL) % BB;
  const int l = blk / (LL * BB);
  const int tid = threadIdx.x;
  const size_t rowF = ((size_t)(l * BB + b) * TT + t) * DD;
  const size_t rowR = ((size_t)(l * BB + b) * TT + (TT - 1 - t)) * DD;
  float v0 = 0.5f * (ob[rowF + tid]       + ob[rowR + tid]);
  float v1 = 0.5f * (ob[rowF + tid + 256] + ob[rowR + tid + 256]);
  __shared__ float s1[256], s2[256];
  s1[tid] = v0 + v1;
  s2[tid] = v0 * v0 + v1 * v1;
  __syncthreads();
  for (int s = 128; s > 0; s >>= 1) {
    if (tid < s) { s1[tid] += s1[tid + s]; s2[tid] += s2[tid + s]; }
    __syncthreads();
  }
  const float mu  = s1[0] * (1.f / DD);
  const float var = s2[0] * (1.f / DD) - mu * mu;
  const float rs  = rsqrtf(var + 1e-5f);
  const float* X = (l == 0) ? x0 : (l == 1) ? x1 : (l == 2) ? x2 : x3;
  const size_t xrow = ((size_t)b * LL + t) * DD;
  const size_t orow = (size_t)l * BB * LL * DD + xrow;
  out[orow + tid]       = (v0 - mu) * rs * g[tid]       + be[tid]       + X[xrow + tid];
  out[orow + tid + 256] = (v1 - mu) * rs * g[tid + 256] + be[tid + 256] + X[xrow + tid + 256];
}

// ------------------------------------------------------------------
extern "C" void kernel_launch(void* const* d_in, const int* in_sizes, int n_in,
                              void* d_out, int out_size, void* d_ws, size_t ws_size,
                              hipStream_t stream)
{
  (void)in_sizes; (void)n_in; (void)out_size; (void)ws_size;
  const float* x0      = (const float*)d_in[0];
  const float* x1      = (const float*)d_in[1];
  const float* x2      = (const float*)d_in[2];
  const float* x3      = (const float*)d_in[3];
  const float* w_in_f  = (const float*)d_in[4];
  const float* conv_w  = (const float*)d_in[5];
  const float* conv_b  = (const float*)d_in[6];
  const float* w_x_f   = (const float*)d_in[7];
  const float* w_dt_f  = (const float*)d_in[8];
  const float* dt_b    = (const float*)d_in[9];
  const float* A_log   = (const float*)d_in[10];
  const float* D_par   = (const float*)d_in[11];
  const float* w_out_f = (const float*)d_in[12];
  const float* ln_g    = (const float*)d_in[13];
  const float* ln_b    = (const float*)d_in[14];

  char* ws = (char*)d_ws;
  size_t off = 0;
  auto alloc = [&](size_t bytes) -> char* {
    char* p = ws + off; off += (bytes + 255) & ~(size_t)255; return p;
  };
  bf16*  w_in  = (bf16*) alloc((size_t)NL * 2 * DI * DD * 2);
  bf16*  w_x   = (bf16*) alloc((size_t)NL * (DTR + 2 * DS) * DI * 2);
  bf16*  w_dt  = (bf16*) alloc((size_t)NL * DI * DTR * 2);
  bf16*  w_out = (bf16*) alloc((size_t)NL * DD * DI * 2);
  bf16*  xd    = (bf16*) alloc((size_t)NL * MM * DD * 2);
  bf16*  xz    = (bf16*) alloc((size_t)NL * MM * 2 * DI * 2);
  bf16*  xc    = (bf16*) alloc((size_t)NL * MM * DI * 2);
  float* xdbl  = (float*)alloc((size_t)NL * MM * (DTR + 2 * DS) * 4);
  bf16*  xdta  = (bf16*) alloc((size_t)NL * MM * DTR * 2);
  float* dtin  = (float*)alloc((size_t)NL * MM * DI * 4);
  bf16*  yb    = (bf16*) alloc((size_t)NL * MM * DI * 2);
  float* ob    = (float*)alloc((size_t)NL * MM * DD * 4);

  auto cvt = [&](const float* src, bf16* dst, int n) {
    k_f32_to_bf16<<<(n + 255) / 256, 256, 0, stream>>>(src, dst, n);
  };
  cvt(w_in_f,  w_in,  NL * 2 * DI * DD);
  cvt(w_x_f,   w_x,   NL * (DTR + 2 * DS) * DI);
  cvt(w_dt_f,  w_dt,  NL * DI * DTR);
  cvt(w_out_f, w_out, NL * DD * DI);

  {
    int n = NL * BB * TT * DD;
    k_build_xd<<<(n + 255) / 256, 256, 0, stream>>>(x0, x1, x2, x3, xd);
  }
  // in_proj: (8192 x 2048) = (8192 x 512) * W^T      per layer
  wmma_gemm<128, 128, true ><<<dim3(MM / 128, (2 * DI) / 128, NL), 256, 0, stream>>>(
      xd, w_in, xz, MM, 2 * DI, DD, DD);
  {
    long n = (long)NL * BB * TT * DI;
    k_conv_silu<<<(unsigned)((n + 255) / 256), 256, 0, stream>>>(xz, conv_w, conv_b, xc);
  }
  // x_proj: (8192 x 64) = (8192 x 1024) * W^T
  wmma_gemm<128, 64, false><<<dim3(MM / 128, 1, NL), 256, 0, stream>>>(
      xc, w_x, xdbl, MM, DTR + 2 * DS, DI, DI);
  {
    int n = NL * MM * DTR;
    k_slice_dt<<<(n + 255) / 256, 256, 0, stream>>>(xdbl, xdta, n);
  }
  // dt_proj: (8192 x 1024) = (8192 x 32) * W^T
  wmma_gemm<128, 128, false><<<dim3(MM / 128, DI / 128, NL), 256, 0, stream>>>(
      xdta, w_dt, dtin, MM, DI, DTR, DTR);
  // fused softplus + selective scan + D-skip + z-gate
  k_scan<<<dim3(NL * BB, DI / 64), 64, 0, stream>>>(
      dtin, dt_b, xdbl, xc, xz, A_log, D_par, yb);
  // out_proj: (8192 x 512) = (8192 x 1024) * W^T
  wmma_gemm<128, 128, false><<<dim3(MM / 128, DD / 128, NL), 256, 0, stream>>>(
      yb, w_out, ob, MM, DD, DI, DI);
  // combine fwd/bwd halves + LayerNorm + residual -> d_out (4 outputs concat)
  k_combine_ln<<<NL * BB * LL, 256, 0, stream>>>(
      ob, x0, x1, x2, x3, ln_g, ln_b, (float*)d_out);
}
